// AttentionConv2d_46265387712524
// MI455X (gfx1250) — matrix operand order repaired and checked by reading, not compiled, not run on hardware
//
#include <hip/hip_runtime.h>
#include <hip/hip_bf16.h>
#include <math.h>

typedef _Float16 half_t;
typedef __attribute__((ext_vector_type(16))) _Float16 v16h;
typedef __attribute__((ext_vector_type(8)))  float    v8f;
typedef __attribute__((ext_vector_type(4)))  unsigned int v4u;

#define BATCH 8
#define CCH   256
#define NPIX  4096
#define RDIM  32

// ---- WMMA wrapper: D = A(16x32 f16) * B(32x16 f16) + C(16x16 f32) ----
__device__ __forceinline__ v8f wmma_f16(v16h a, v16h b, v8f c) {
  return __builtin_amdgcn_wmma_f32_16x16x32_f16(
      /*neg_a=*/false, a, /*neg_b=*/false, b,
      /*c_mod=*/(short)0, c, /*reuse_a=*/false, /*reuse_b=*/false);
}

// A fragment from f16 row-major [16 x >=32], ld in halfs.
// ISA layout: lane l (m=l&15, hi=l>>4): e[0..7]=A[m, hi*8+0..7], e[8..15]=A[m, 16+hi*8+0..7]
__device__ __forceinline__ v16h load_a_f16(const half_t* base, int ld, int lane) {
  int m = lane & 15, hi = lane >> 4;
  const half_t* p0 = base + (size_t)m * ld + hi * 8;
  const half_t* p1 = p0 + 16;
  union { v4u u[2]; v16h h; } cvt;
  cvt.u[0] = *reinterpret_cast<const v4u*>(p0);
  cvt.u[1] = *reinterpret_cast<const v4u*>(p1);
  return cvt.h;
}

// A fragment from f32 row-major weights, converted to f16.
__device__ __forceinline__ v16h load_a_f32cvt(const float* base, int ld, int lane) {
  int m = lane & 15, hi = lane >> 4;
  const float* p0 = base + (size_t)m * ld + hi * 8;
  const float* p1 = p0 + 16;
  v16h a;
#pragma unroll
  for (int j = 0; j < 8; ++j) a[j] = (_Float16)p0[j];
#pragma unroll
  for (int j = 0; j < 8; ++j) a[j + 8] = (_Float16)p1[j];
  return a;
}

// B fragment: columns stored contiguously (layout BT[n][k], ld in halfs).
// ISA layout: lane l (n=l&15, hi=l>>4): e[0..7]=B[hi*8+0..7, n], e[8..15]=B[16+hi*8+0..7, n]
__device__ __forceinline__ v16h load_b_f16(const half_t* base, int ld, int lane) {
  int n = lane & 15, hi = lane >> 4;
  const half_t* p = base + (size_t)n * ld + hi * 8;
  union { v4u u[2]; v16h h; } cvt;
  cvt.u[0] = *reinterpret_cast<const v4u*>(p);
  cvt.u[1] = *reinterpret_cast<const v4u*>(p + 16);
  return cvt.h;
}

// ============================================================================
// Kernel 1: projections.  qT,kT: [B][N][32] f16 (transposed), vh: [B][256][N] f16
// grid: (N/64, B), 256 threads (8 waves). 80 output tiles of 16x16 per block.
// ============================================================================
__global__ __launch_bounds__(256) void proj_kernel(
    const float* __restrict__ x,
    const float* __restrict__ Wq, const float* __restrict__ bq,
    const float* __restrict__ Wk, const float* __restrict__ bk,
    const float* __restrict__ Wv, const float* __restrict__ bv,
    half_t* __restrict__ qT, half_t* __restrict__ kT, half_t* __restrict__ vh)
{
  const int b  = blockIdx.y;
  const int n0 = blockIdx.x * 64;
  __shared__ half_t xs[64][264];                 // [pixel n][channel c], f16, padded

  const float* xb = x + (size_t)b * CCH * NPIX;
  const int t = threadIdx.x;
  const int j = t & 63;                          // pixel within tile
  for (int c = (t >> 6); c < CCH; c += 4)
    xs[j][c] = (_Float16)xb[(size_t)c * NPIX + n0 + j];
  __syncthreads();

  const int w = t >> 5, lane = t & 31, hi = lane >> 4;
  for (int T = w; T < 80; T += 8) {
    const int rt = T >> 2, ct = T & 3;
    const float* Wm; const float* bias; int o0; half_t* outp; bool transT;
    if (rt < 2)      { Wm = Wq; bias = bq; o0 = rt * 16;        outp = qT; transT = true;  }
    else if (rt < 4) { Wm = Wk; bias = bk; o0 = (rt - 2) * 16;  outp = kT; transT = true;  }
    else             { Wm = Wv; bias = bv; o0 = (rt - 4) * 16;  outp = vh; transT = false; }

    v8f acc = {};
#pragma unroll
    for (int k0 = 0; k0 < CCH; k0 += 32) {
      v16h a = load_a_f32cvt(Wm + (size_t)o0 * CCH + k0, CCH, lane);
      v16h bf = load_b_f16(&xs[ct * 16][k0], 264, lane);
      acc = wmma_f16(a, bf, acc);
    }

    const int n = n0 + ct * 16 + (lane & 15);
    if (transT) {
      // pack 8 consecutive output rows (o0+8*hi .. +7) -> one 16B store into qT/kT[n][*]
      union { v4u u; _Float16 h[8]; } pk;
#pragma unroll
      for (int r = 0; r < 8; ++r)
        pk.h[r] = (_Float16)(acc[r] + bias[o0 + r + 8 * hi]);
      *reinterpret_cast<v4u*>(outp + ((size_t)b * NPIX + n) * RDIM + o0 + 8 * hi) = pk.u;
    } else {
#pragma unroll
      for (int r = 0; r < 8; ++r) {
        const int o = o0 + r + 8 * hi;
        outp[((size_t)b * CCH + o) * NPIX + n] = (_Float16)(acc[r] + bias[o]);
      }
    }
  }
}

// ============================================================================
// Kernel 2: softmax stats over n for each column m:  max_n S[n,m], sum_n exp(S-max)
// S[n,m] = q[:,n] . k[:,m].  grid: (N/64, B), 8 waves split the n range (512 each).
// ============================================================================
__global__ __launch_bounds__(256) void stats_kernel(
    const half_t* __restrict__ qT, const half_t* __restrict__ kT,
    float* __restrict__ maxv, float* __restrict__ sumv)
{
  const int b = blockIdx.y, m0 = blockIdx.x * 64;
  const int t = threadIdx.x, w = t >> 5, lane = t & 31;
  __shared__ float wmaxs[8][64];
  __shared__ float wsums[8][64];

  const half_t* qTb = qT + (size_t)b * NPIX * RDIM;
  const half_t* kTb = kT + (size_t)b * NPIX * RDIM;

  v16h bf[4];                                    // loop-invariant k fragments
#pragma unroll
  for (int ct = 0; ct < 4; ++ct)
    bf[ct] = load_b_f16(kTb + (size_t)(m0 + ct * 16) * RDIM, RDIM, lane);

  float rmax[4], rsum[4];
#pragma unroll
  for (int ct = 0; ct < 4; ++ct) { rmax[ct] = -INFINITY; rsum[ct] = 0.f; }

  for (int n0 = w * 512; n0 < (w + 1) * 512; n0 += 16) {
    v16h a = load_a_f16(qTb + (size_t)n0 * RDIM, RDIM, lane);
#pragma unroll
    for (int ct = 0; ct < 4; ++ct) {
      v8f s = {};
      s = wmma_f16(a, bf[ct], s);
      float tm = s[0];
#pragma unroll
      for (int r = 1; r < 8; ++r) tm = fmaxf(tm, s[r]);
      const float nm = fmaxf(rmax[ct], tm);
      float part = 0.f;
#pragma unroll
      for (int r = 0; r < 8; ++r) part += __expf(s[r] - nm);
      rsum[ct] = rsum[ct] * __expf(rmax[ct] - nm) + part;
      rmax[ct] = nm;
    }
  }
  // merge lane pairs (lane l and l^16 hold the two row-halves of the same column)
#pragma unroll
  for (int ct = 0; ct < 4; ++ct) {
    const float om = __shfl_xor(rmax[ct], 16);
    const float os = __shfl_xor(rsum[ct], 16);
    const float nm = fmaxf(rmax[ct], om);
    rsum[ct] = rsum[ct] * __expf(rmax[ct] - nm) + os * __expf(om - nm);
    rmax[ct] = nm;
  }
  if (lane < 16) {
#pragma unroll
    for (int ct = 0; ct < 4; ++ct) {
      wmaxs[w][ct * 16 + lane] = rmax[ct];
      wsums[w][ct * 16 + lane] = rsum[ct];
    }
  }
  __syncthreads();
  if (t < 64) {
    float m = -INFINITY;
#pragma unroll
    for (int wv = 0; wv < 8; ++wv) m = fmaxf(m, wmaxs[wv][t]);
    float s = 0.f;
#pragma unroll
    for (int wv = 0; wv < 8; ++wv) s += wsums[wv][t] * __expf(wmaxs[wv][t] - m);
    maxv[(size_t)b * NPIX + m0 + t] = m;
    sumv[(size_t)b * NPIX + m0 + t] = s;
  }
}

// ============================================================================
// Kernel 3: O[c,m] = sum_n v[c,n]*exp(S[n,m]-max_m);  out = x + gamma*O/sum_m
// grid: (N/64, B).  n-step = 64, double-buffered P tile, ONE barrier/iter.
// Wave w: C rows [32w, 32w+32); S tiles: nsub = w>>1, ct in {2*(w&1), 2*(w&1)+1}.
// ============================================================================
__global__ __launch_bounds__(256) void attn_kernel(
    const float* __restrict__ x,
    const half_t* __restrict__ qT, const half_t* __restrict__ kT,
    const half_t* __restrict__ vh,
    const float* __restrict__ maxv, const float* __restrict__ sumv,
    const float* __restrict__ gamma, float* __restrict__ out)
{
  const int b = blockIdx.y, m0 = blockIdx.x * 64;
  const int t = threadIdx.x, w = t >> 5, lane = t & 31, hi = lane >> 4;
  __shared__ half_t Pt[2][64][72];               // double-buffered [m][n], rows 144B

  const half_t* qTb = qT + (size_t)b * NPIX * RDIM;
  const half_t* kTb = kT + (size_t)b * NPIX * RDIM;
  const half_t* vb  = vh + (size_t)b * CCH * NPIX;
  const float g = gamma[0];

  float mx[4], fs[4];                            // per-column stats for this lane
#pragma unroll
  for (int ct = 0; ct < 4; ++ct) {
    const int col = m0 + ct * 16 + (lane & 15);
    mx[ct] = maxv[(size_t)b * NPIX + col];
    fs[ct] = g / sumv[(size_t)b * NPIX + col];
  }

  const int nsub = w >> 1;                       // 0..3: n-subtile within 64-row step
  const int sct0 = (w & 1) * 2;                  // this wave's m-coltile pair
  v16h bS[2];
#pragma unroll
  for (int i = 0; i < 2; ++i)
    bS[i] = load_b_f16(kTb + (size_t)(m0 + (sct0 + i) * 16) * RDIM, RDIM, lane);

  v8f acc[2][4];
#pragma unroll
  for (int rt = 0; rt < 2; ++rt)
#pragma unroll
    for (int ct = 0; ct < 4; ++ct) acc[rt][ct] = (v8f){};

  const int c0 = w * 32;

  // --- S tile producer: two 16x16 tiles -> exp -> packed 16B LDS stores ---
  auto produceS = [&](int n0s, int buf) {
    v16h aS = load_a_f16(qTb + (size_t)(n0s + nsub * 16) * RDIM, RDIM, lane);
#pragma unroll
    for (int i = 0; i < 2; ++i) {
      const int ct = sct0 + i;
      v8f s = {};
      s = wmma_f16(aS, bS[i], s);
      union { v4u u; _Float16 h[8]; } pk;
#pragma unroll
      for (int r = 0; r < 8; ++r)
        pk.h[r] = (_Float16)__expf(s[r] - mx[ct]);
      // rows r..r+7 are consecutive n -> one 16B packed LDS store
      *reinterpret_cast<v4u*>(&Pt[buf][ct * 16 + (lane & 15)][nsub * 16 + 8 * hi]) = pk.u;
    }
  };

  produceS(0, 0);
  __syncthreads();

  for (int n0 = 0; n0 < NPIX; n0 += 64) {
    const int buf = (n0 >> 6) & 1;
    // produce next iteration's P into the other buffer (overlaps with PV below)
    if (n0 + 64 < NPIX) produceS(n0 + 64, buf ^ 1);

    // --- PV: acc[rt][ct] += v[c-tile, n0:64] * P[n0:64, m-tile] ---
#pragma unroll
    for (int ks = 0; ks < 2; ++ks) {
      v16h av0 = load_a_f16(vb + (size_t)c0 * NPIX + n0 + ks * 32, NPIX, lane);
      v16h av1 = load_a_f16(vb + (size_t)(c0 + 16) * NPIX + n0 + ks * 32, NPIX, lane);
#pragma unroll
      for (int ct = 0; ct < 4; ++ct) {
        v16h bp = load_b_f16(&Pt[buf][ct * 16][ks * 32], 72, lane);
        acc[0][ct] = wmma_f16(av0, bp, acc[0][ct]);
        acc[1][ct] = wmma_f16(av1, bp, acc[1][ct]);
      }
    }
    __syncthreads();  // next buffer fully written; this buffer fully read
  }

  // epilogue: out = x + (gamma/sum) * O
#pragma unroll
  for (int rt = 0; rt < 2; ++rt)
#pragma unroll
    for (int ct = 0; ct < 4; ++ct)
#pragma unroll
      for (int r = 0; r < 8; ++r) {
        const int c = c0 + rt * 16 + r + 8 * hi;
        const int m = m0 + ct * 16 + (lane & 15);
        const size_t idx = ((size_t)b * CCH + c) * NPIX + m;
        out[idx] = x[idx] + fs[ct] * acc[rt][ct][r];
      }
}

// ============================================================================
extern "C" void kernel_launch(void* const* d_in, const int* in_sizes, int n_in,
                              void* d_out, int out_size, void* d_ws, size_t ws_size,
                              hipStream_t stream) {
  (void)in_sizes; (void)n_in; (void)out_size; (void)ws_size;
  const float* x     = (const float*)d_in[0];
  const float* Wq    = (const float*)d_in[1];
  const float* bq    = (const float*)d_in[2];
  const float* Wk    = (const float*)d_in[3];
  const float* bk    = (const float*)d_in[4];
  const float* Wv    = (const float*)d_in[5];
  const float* bv    = (const float*)d_in[6];
  const float* gamma = (const float*)d_in[7];

  char* ws = (char*)d_ws;
  half_t* qT   = (half_t*)(ws);                               // 8*4096*32*2  = 2 MB
  half_t* kT   = (half_t*)(ws + (size_t)2  * 1024 * 1024);    // 2 MB
  half_t* vh   = (half_t*)(ws + (size_t)4  * 1024 * 1024);    // 8*256*4096*2 = 16 MB
  float*  maxv = (float*) (ws + (size_t)20 * 1024 * 1024);    // 128 KB
  float*  sumv = maxv + (size_t)BATCH * NPIX;                 // 128 KB
  float*  out  = (float*)d_out;

  dim3 grid(NPIX / 64, BATCH);
  proj_kernel <<<grid, 256, 0, stream>>>(x, Wq, bq, Wk, bk, Wv, bv, qT, kT, vh);
  stats_kernel<<<grid, 256, 0, stream>>>(qT, kT, maxv, sumv);
  attn_kernel <<<grid, 256, 0, stream>>>(x, qT, kT, vh, maxv, sumv, gamma, out);
}
